// VectorQuantize_63763084476534
// MI455X (gfx1250) — matrix operand order repaired and checked by reading, not compiled, not run on hardware
//
#include <hip/hip_runtime.h>
#include <hip/hip_bf16.h>

typedef __attribute__((ext_vector_type(16))) __bf16 v16bf;
typedef __attribute__((ext_vector_type(8)))  float  v8f;
typedef __attribute__((ext_vector_type(8)))  unsigned short u16x8;
typedef __attribute__((ext_vector_type(4)))  int    v4i;

#define T_TOK 32768
#define DIM   256
#define KCODE 8192
#define DECAY 0.8f
#define OMD   0.2f
#define EPSV  1e-5f

union ABFrag { u16x8 h[2]; v16bf v; };

#if __has_builtin(__builtin_amdgcn_global_load_async_to_lds_b128)
#define HAVE_ASYNC_LDS 1
#else
#define HAVE_ASYNC_LDS 0
#endif

#if HAVE_ASYNC_LDS
typedef __attribute__((address_space(1))) v4i* g_v4i_p;
typedef __attribute__((address_space(3))) v4i* l_v4i_p;
#if __has_builtin(__builtin_amdgcn_s_wait_asynccnt)
#define WAIT_ASYNC(n) __builtin_amdgcn_s_wait_asynccnt(n)
#else
#define WAIT_ASYNC(n) asm volatile("s_wait_asynccnt %0" ::"i"(n))
#endif
#else
#define WAIT_ASYNC(n)
#endif

__device__ __forceinline__ unsigned short f2bf(float f) {
    unsigned int u = __float_as_uint(f);
    unsigned int r = (u + 0x7FFFu + ((u >> 16) & 1u)) >> 16;
    return (unsigned short)r;
}

// Stage one 64-code x 256-dim bf16 tile (2048 uint4) into LDS.
__device__ __forceinline__ void stage_tile(const uint4* __restrict__ src, uint4* dst, int tid) {
#if HAVE_ASYNC_LDS
#pragma unroll
    for (int i = 0; i < 8; ++i) {
        int idx = tid + i * 256;
        __builtin_amdgcn_global_load_async_to_lds_b128(
            (g_v4i_p)(unsigned long long)(const void*)(src + idx),
            (l_v4i_p)(unsigned)(unsigned long long)(const void*)(dst + idx),
            0, 0);
    }
#else
#pragma unroll
    for (int i = 0; i < 8; ++i) dst[tid + i * 256] = src[tid + i * 256];
#endif
}

// ---- K0: embed -> bf16 copy + bias[k] = -0.5*||e_k||^2 ----
__global__ __launch_bounds__(256) void vq_prep_embed(const float* __restrict__ embed,
                                                     unsigned short* __restrict__ eb,
                                                     float* __restrict__ bias) {
    __shared__ float sm[256];
    int k = blockIdx.x, d = threadIdx.x;
    float v = embed[(size_t)k * DIM + d];
    eb[(size_t)k * DIM + d] = f2bf(v);
    sm[d] = v * v;
    __syncthreads();
    for (int s = 128; s > 0; s >>= 1) {
        if (d < s) sm[d] += sm[d + s];
        __syncthreads();
    }
    if (d == 0) bias[k] = -0.5f * sm[0];
}

// ---- K1: x -> bf16 copy ----
__global__ __launch_bounds__(256) void vq_prep_x(const float* __restrict__ x,
                                                 unsigned short* __restrict__ xb, int n) {
    for (int i = blockIdx.x * blockDim.x + threadIdx.x; i < n; i += gridDim.x * blockDim.x)
        xb[i] = f2bf(x[i]);
}

// ---- K2: WMMA distance GEMM + running argmax, async double-buffered B tiles ----
// 256 blocks x 256 threads (8 waves). Wave w owns token tile mTile = blk*8+w.
__global__ __launch_bounds__(256) void vq_gemm_argmax(const unsigned short* __restrict__ xb,
                                                      const unsigned short* __restrict__ eb,
                                                      const float* __restrict__ bias,
                                                      int* __restrict__ ind,
                                                      float* __restrict__ indf) {
    __shared__ uint4 tileB[2][2048]; // 2 x 32 KB: 64 codes x 256 dims bf16 each
    const int tid = threadIdx.x;
    const int lane = tid & 31;
    const int wave = tid >> 5;
    const int m0 = (blockIdx.x * 8 + wave) * 16;
    const int row = m0 + (lane & 15);
    const int aoff = (lane < 16) ? 0 : 8;   // A 16-bit layout: halves own K {0..7,16..23} / {8..15,24..31}

    // Persistent A fragments covering D=256 (8 chunks of K=32)
    v16bf a[8];
#pragma unroll
    for (int kc = 0; kc < 8; ++kc) {
        const u16x8* p = (const u16x8*)(xb + (size_t)row * DIM + kc * 32 + aoff);
        ABFrag f;
        f.h[0] = p[0];   // K = kc*32 + aoff + 0..7
        f.h[1] = p[2];   // K = kc*32 + aoff + 16..23
        a[kc] = f.v;
    }

    float best[8];
    int bidx[8];
#pragma unroll
    for (int i = 0; i < 8; ++i) { best[i] = -3.0e38f; bidx[i] = 0; }

    const int kOffB = (lane < 16) ? 0 : 8; // B layout: halves own K {0..7,16..23} / {8..15,24..31} per chunk

    // Prologue: stage tile 0 into buffer 0
    stage_tile((const uint4*)eb, &tileB[0][0], tid);

    for (int nb = 0; nb < KCODE; nb += 64) {
        const int b = (nb >> 6) & 1;
        __syncthreads(); // all waves done computing on buffer b^1 (previous iteration)
        if (nb + 64 < KCODE) {
            stage_tile((const uint4*)(eb + (size_t)(nb + 64) * DIM), &tileB[b ^ 1][0], tid);
            WAIT_ASYNC(8);  // leave only the just-issued stage outstanding
        } else {
            WAIT_ASYNC(0);
        }
        __syncthreads(); // buffer b fully populated across all waves

        const unsigned short* tb = (const unsigned short*)&tileB[b][0];
#pragma unroll
        for (int sub = 0; sub < 4; ++sub) {
            const int lcol = sub * 16 + (lane & 15);     // code row within LDS tile
            const int code = nb + lcol;
            const float bv = bias[code];                 // -0.5*||e||^2 (L2-resident)
            const u16x8* bp = (const u16x8*)(tb + lcol * DIM + kOffB);
            // Load all 8 B fragments first: one LDS clause, then 8 back-to-back WMMAs
            v16bf bb[8];
#pragma unroll
            for (int kc = 0; kc < 8; ++kc) {
                ABFrag bf;
                bf.h[0] = bp[kc * 4];       // K = kc*32 + kOffB + 0..7
                bf.h[1] = bp[kc * 4 + 2];   // K = kc*32 + kOffB + 16..23
                bb[kc] = bf.v;
            }
            v8f c = {bv, bv, bv, bv, bv, bv, bv, bv};
#pragma unroll
            for (int kc = 0; kc < 8; ++kc)
                c = __builtin_amdgcn_wmma_f32_16x16x32_bf16(false, a[kc], false, bb[kc],
                                                            (short)0, c, false, false);
#pragma unroll
            for (int i = 0; i < 8; ++i) {
                if (c[i] > best[i]) { best[i] = c[i]; bidx[i] = code; }
            }
        }
    }

    // Argmax reduce across the 16 lanes holding the same rows (xor stays inside the half)
#pragma unroll
    for (int i = 0; i < 8; ++i) {
        float s = best[i];
        int id = bidx[i];
#pragma unroll
        for (int m = 8; m >= 1; m >>= 1) {
            float so = __shfl_xor(s, m, 32);
            int io = __shfl_xor(id, m, 32);
            if (so > s || (so == s && io < id)) { s = so; id = io; }
        }
        best[i] = s; bidx[i] = id;
    }
    if (lane == 0) {
#pragma unroll
        for (int i = 0; i < 8; ++i) { ind[m0 + i] = bidx[i]; indf[m0 + i] = (float)bidx[i]; }
    }
    if (lane == 16) {
#pragma unroll
        for (int i = 0; i < 8; ++i) { ind[m0 + 8 + i] = bidx[i]; indf[m0 + 8 + i] = (float)bidx[i]; }
    }
}

// ---- K3: gather quantize (= straight-through forward value) + loss partials ----
__global__ __launch_bounds__(256) void vq_gather_loss(const int* __restrict__ ind,
                                                      const float* __restrict__ x,
                                                      const float* __restrict__ embed,
                                                      float* __restrict__ outQ,
                                                      float* __restrict__ lossPart) {
    __shared__ float sm[256];
    int b = blockIdx.x;
    float acc = 0.f;
    for (int it = 0; it < 32; ++it) {
        int t = b * 32 + it;
        int idx = ind[t];
        float xv = x[(size_t)t * DIM + threadIdx.x];
        float q = embed[(size_t)idx * DIM + threadIdx.x];
        outQ[(size_t)t * DIM + threadIdx.x] = q;
        float d = q - xv;
        acc += d * d;
    }
    sm[threadIdx.x] = acc;
    __syncthreads();
    for (int s = 128; s > 0; s >>= 1) {
        if (threadIdx.x < s) sm[threadIdx.x] += sm[threadIdx.x + s];
        __syncthreads();
    }
    if (threadIdx.x == 0) lossPart[b] = sm[0];
}

__global__ __launch_bounds__(256) void vq_loss_final(const float* __restrict__ lossPart,
                                                     float* __restrict__ outLoss) {
    __shared__ float sm[256];
    float a = 0.f;
    for (int i = threadIdx.x; i < 1024; i += 256) a += lossPart[i];
    sm[threadIdx.x] = a;
    __syncthreads();
    for (int s = 128; s > 0; s >>= 1) {
        if (threadIdx.x < s) sm[threadIdx.x] += sm[threadIdx.x + s];
        __syncthreads();
    }
    if (threadIdx.x == 0) outLoss[0] = sm[0] / (float)((size_t)T_TOK * DIM);
}

// ---- K4: deterministic scatter stats: one block per code scans the index array ----
__global__ __launch_bounds__(256) void vq_stats(const int* __restrict__ ind,
                                                const float* __restrict__ x,
                                                float* __restrict__ bins,
                                                float* __restrict__ esum) {
    __shared__ int sIdx[4096]; // 16 KB chunk of indices
    const int k = blockIdx.x;
    const int d = threadIdx.x;
    float acc = 0.f;
    int cnt = 0;
    for (int c0 = 0; c0 < T_TOK; c0 += 4096) {
        __syncthreads();
        for (int i = threadIdx.x; i < 4096; i += 256) sIdx[i] = ind[c0 + i];
        __syncthreads();
        const int4* s4 = (const int4*)sIdx;
        for (int t4 = 0; t4 < 1024; ++t4) {
            int4 q = s4[t4];
            int base = c0 + t4 * 4;
            if (q.x == k) { acc += x[(size_t)(base + 0) * DIM + d]; cnt++; }
            if (q.y == k) { acc += x[(size_t)(base + 1) * DIM + d]; cnt++; }
            if (q.z == k) { acc += x[(size_t)(base + 2) * DIM + d]; cnt++; }
            if (q.w == k) { acc += x[(size_t)(base + 3) * DIM + d]; cnt++; }
        }
    }
    esum[(size_t)k * DIM + d] = acc;
    if (d == 0) bins[k] = (float)cnt;
}

// ---- K5: EMA updates ----
__global__ __launch_bounds__(256) void vq_ema_embed(const float* __restrict__ ea,
                                                    const float* __restrict__ esum,
                                                    float* __restrict__ outNEA, int n) {
    for (int i = blockIdx.x * blockDim.x + threadIdx.x; i < n; i += gridDim.x * blockDim.x)
        outNEA[i] = ea[i] * DECAY + OMD * esum[i];
}

__global__ __launch_bounds__(256) void vq_ema_cs(const float* __restrict__ cs,
                                                 const float* __restrict__ bins,
                                                 float* __restrict__ outNCS) {
    int i = blockIdx.x * blockDim.x + threadIdx.x;
    if (i < KCODE) outNCS[i] = cs[i] * DECAY + OMD * bins[i];
}

__global__ __launch_bounds__(256) void vq_total(const float* __restrict__ ncs,
                                                float* __restrict__ totalP) {
    __shared__ float sm[256];
    float a = 0.f;
    for (int i = threadIdx.x; i < KCODE; i += 256) a += ncs[i];
    sm[threadIdx.x] = a;
    __syncthreads();
    for (int s = 128; s > 0; s >>= 1) {
        if (threadIdx.x < s) sm[threadIdx.x] += sm[threadIdx.x + s];
        __syncthreads();
    }
    if (threadIdx.x == 0) totalP[0] = sm[0];
}

__global__ __launch_bounds__(256) void vq_newembed(const float* __restrict__ nea,
                                                   const float* __restrict__ ncs,
                                                   const float* __restrict__ totalP,
                                                   float* __restrict__ outNE, int n) {
    float tot = totalP[0];
    float denom = tot + (float)KCODE * EPSV;
    for (int i = blockIdx.x * blockDim.x + threadIdx.x; i < n; i += gridDim.x * blockDim.x) {
        int k = i >> 8;
        float smoothed = (ncs[k] + EPSV) / denom * tot;
        outNE[i] = nea[i] / smoothed;
    }
}

extern "C" void kernel_launch(void* const* d_in, const int* in_sizes, int n_in,
                              void* d_out, int out_size, void* d_ws, size_t ws_size,
                              hipStream_t stream) {
    const float* x     = (const float*)d_in[0]; // [8,4096,256]
    const float* embed = (const float*)d_in[1]; // [8192,256]
    const float* cs    = (const float*)d_in[2]; // [8192]
    const float* ea    = (const float*)d_in[3]; // [8192,256]

    float* out      = (float*)d_out;
    float* outQ     = out;                         // 8388608
    float* outInd   = out + 8388608;               // 32768
    float* outLoss  = out + 8421376;               // 1
    float* outNE    = out + 8421377;               // 2097152
    float* outNCS   = out + 10518529;              // 8192
    float* outNEA   = out + 10526721;              // 2097152

    char* ws = (char*)d_ws;
    unsigned short* xb   = (unsigned short*)(ws + 0);          // 16 MB
    unsigned short* eb   = (unsigned short*)(ws + 16777216);   //  4 MB
    float*          bias = (float*)(ws + 20971520);            // 32 KB
    int*            ind  = (int*)(ws + 21004288);              // 128 KB
    float*          bins = (float*)(ws + 21135360);            // 32 KB
    float*          esum = (float*)(ws + 21168128);            //  8 MB
    float*          lossPart = (float*)(ws + 29556736);        //  4 KB
    float*          totalP   = (float*)(ws + 29560832);        //  4 B

    vq_prep_embed<<<KCODE, 256, 0, stream>>>(embed, eb, bias);
    vq_prep_x<<<4096, 256, 0, stream>>>(x, xb, T_TOK * DIM);
    vq_gemm_argmax<<<256, 256, 0, stream>>>(xb, eb, bias, ind, outInd);
    vq_gather_loss<<<1024, 256, 0, stream>>>(ind, x, embed, outQ, lossPart);
    vq_loss_final<<<1, 256, 0, stream>>>(lossPart, outLoss);
    vq_stats<<<KCODE, 256, 0, stream>>>(ind, x, bins, esum);
    vq_ema_embed<<<2048, 256, 0, stream>>>(ea, esum, outNEA, KCODE * DIM);
    vq_ema_cs<<<32, 256, 0, stream>>>(cs, bins, outNCS);
    vq_total<<<1, 256, 0, stream>>>(outNCS, totalP);
    vq_newembed<<<2048, 256, 0, stream>>>(outNEA, outNCS, totalP, outNE, KCODE * DIM);
}
